// ESN_21010980012342
// MI455X (gfx1250) — matrix operand order repaired and checked by reading, not compiled, not run on hardware
//
#include <hip/hip_runtime.h>
#include <cmath>

// ---- problem constants (from reference) ----
#define ESN_DIM   900
#define ESN_RT    57          // 57 row tiles of 16   (912 >= 900)
#define ESN_KTP   32          // K tiles padded 29->32 (1024 >= 900), 8 per wave
#define ESN_XPAD  1024        // padded state length
#define ESN_T     31000       // TRAIN_LEN + DISCARD
#define ESN_SKIP  1001        // first emitted state index (DISCARD+1)
#define ESN_COLS  29999       // output columns

typedef __bf16 v16bf __attribute__((ext_vector_type(16)));
typedef float  v8f   __attribute__((ext_vector_type(8)));

// ---- workspace layout (bytes) ----
//  A_sw : 57*32*512 bf16 swizzled weights   = 1,867,776
//  xg   : 928 fp32 state                    @ 1,867,776 (3712 B)
//  xbf  : 2 x 1024 bf16 state (double buf)  @ 1,871,488 (4096 B)
//  bar  : 2 x u32 grid barrier              @ 1,875,584
#define WS_XG   1867776
#define WS_XBF  1871488
#define WS_BAR  1875584

__global__ void esn_init(float* xg, __bf16* xbf, unsigned* bar) {
  int i = blockIdx.x * blockDim.x + threadIdx.x;
  if (i < 2 * ESN_XPAD) xbf[i] = (__bf16)0.0f;  // both buffers + K padding = 0
  if (i < 928)          xg[i]  = 0.0f;
  if (i < 2)            bar[i] = 0u;
}

// Pre-swizzle 0.9*W into this code's 16x32 A-operand lane layout (contiguous k, same order as B):
//  lanes 0-15 : M=lane,    VGPR v: K=2v,2v+1
//  lanes 16-31: M=lane-16, VGPR v: K=2v+16,2v+17
// Stored so each lane's operand is one contiguous 32B chunk. K zero-padded.
__global__ void esn_swizzleW(const float* __restrict__ W, __bf16* __restrict__ A) {
  int idx = blockIdx.x * blockDim.x + threadIdx.x;
  if (idx >= ESN_RT * ESN_KTP * 512) return;
  int e    = idx & 15;
  int lane = (idx >> 4) & 31;
  int kt   = (idx >> 9) & 31;             // KTP = 32 -> mask
  int rt   = idx / (ESN_KTP * 512);
  int grp  = lane >> 4;
  int m    = lane & 15;
  int v = e >> 1, h = e & 1;
  int kl = 2 * v + h + (grp ? 16 : 0);
  int row = rt * 16 + m;
  int k   = kt * 32 + kl;
  float val = (row < ESN_DIM && k < ESN_DIM) ? 0.9f * W[row * ESN_DIM + k] : 0.0f;
  A[idx] = (__bf16)val;
}

// Persistent cooperative ESN kernel: 57 WGs x 4 wave32.
// Weights live in VGPRs for the whole kernel; each step streams only the 2KB
// state vector. A sched_barrier pins ALL 8 B-loads ahead of the WMMA chain
// so their L2 latencies overlap (one round-trip per step, not eight).
__global__ __launch_bounds__(128) void esn_persist(
    const __bf16* __restrict__ A, const float* __restrict__ Win,
    const float* __restrict__ u, float* xg, __bf16* xbf,
    unsigned* bar, float* __restrict__ out)
{
  const int rt   = blockIdx.x;
  const int lane = threadIdx.x & 31;
  const int wid  = threadIdx.x >> 5;
  __shared__ float red[4][16];

  // Preload this wave's 8 contiguous K-tiles of A into registers.
  const __bf16* Abase = A + (size_t)rt * (ESN_KTP * 512)
                          + (size_t)(wid * 8) * 512 + lane * 16;
  v16bf areg[8];
#pragma unroll
  for (int i = 0; i < 8; ++i)
    areg[i] = *(const v16bf*)(Abase + i * 512);

  // Per-lane base into the broadcast state vector (lane group selects K half)
  const int xoff = (wid * 8) * 32 + (lane & 16);
  unsigned gen = 0;   // barrier generation (thread 0 only)

  for (int t = 0; t < ESN_T; ++t) {
    const __bf16* xr = xbf + (t & 1) * ESN_XPAD + xoff;

    // Issue ALL state-vector loads first; the sched_barrier stops the
    // scheduler from sinking them back between the WMMAs.
    v16bf breg[8];
#pragma unroll
    for (int i = 0; i < 8; ++i)
      breg[i] = *(const v16bf*)(xr + i * 32);
    __builtin_amdgcn_sched_barrier(0);

    v8f c0 = {}, c1 = {};
#pragma unroll
    for (int i = 0; i < 8; i += 2) {
      c0 = __builtin_amdgcn_wmma_f32_16x16x32_bf16(false, areg[i + 0], false,
                                                   breg[i + 0], (short)0, c0,
                                                   false, false);
      c1 = __builtin_amdgcn_wmma_f32_16x16x32_bf16(false, areg[i + 1], false,
                                                   breg[i + 1], (short)0, c1,
                                                   false, false);
    }
    // All B columns equal -> lane 0 holds y[M=0..7], lane 16 holds y[M=8..15]
    if (lane == 0) {
#pragma unroll
      for (int r = 0; r < 8; ++r) red[wid][r] = c0[r] + c1[r];
    } else if (lane == 16) {
#pragma unroll
      for (int r = 0; r < 8; ++r) red[wid][8 + r] = c0[r] + c1[r];
    }
    __syncthreads();
    if (wid == 0 && lane < 16) {
      int row = rt * 16 + lane;
      if (row < ESN_DIM) {
        float y   = red[0][lane] + red[1][lane] + red[2][lane] + red[3][lane];
        float pre = y + Win[(size_t)row * ESN_DIM] * u[t];   // W_in[:,0]
        float xn  = 0.3f * xg[row] + 0.7f * tanhf(pre);
        xg[row] = xn;
        xbf[((t + 1) & 1) * ESN_XPAD + row] = (__bf16)xn;    // write buffer
        if (t >= ESN_SKIP)
          out[(size_t)row * ESN_COLS + (t - ESN_SKIP)] = xn;
      }
    }
    __syncthreads();
    // device-wide sense-reversing barrier (agent-scope atomics)
    if (threadIdx.x == 0) {
      __threadfence();
      unsigned prev = __hip_atomic_fetch_add(&bar[0], 1u, __ATOMIC_ACQ_REL,
                                             __HIP_MEMORY_SCOPE_AGENT);
      if (prev == (unsigned)gridDim.x - 1u) {
        __hip_atomic_store(&bar[0], 0u, __ATOMIC_RELAXED, __HIP_MEMORY_SCOPE_AGENT);
        __hip_atomic_fetch_add(&bar[1], 1u, __ATOMIC_RELEASE, __HIP_MEMORY_SCOPE_AGENT);
      } else {
        while (__hip_atomic_load(&bar[1], __ATOMIC_ACQUIRE,
                                 __HIP_MEMORY_SCOPE_AGENT) <= gen) {
          __builtin_amdgcn_s_sleep(1);
        }
      }
      ++gen;
    }
    __syncthreads();
  }
}

extern "C" void kernel_launch(void* const* d_in, const int* in_sizes, int n_in,
                              void* d_out, int out_size, void* d_ws, size_t ws_size,
                              hipStream_t stream) {
  (void)in_sizes; (void)n_in; (void)out_size; (void)ws_size;
  const float* W   = (const float*)d_in[0];
  const float* Win = (const float*)d_in[1];
  const float* u   = (const float*)d_in[2];
  float* out = (float*)d_out;
  char*  ws  = (char*)d_ws;
  __bf16*   A   = (__bf16*)ws;
  float*    xg  = (float*)(ws + WS_XG);
  __bf16*   xbf = (__bf16*)(ws + WS_XBF);
  unsigned* bar = (unsigned*)(ws + WS_BAR);

  esn_init<<<8, 256, 0, stream>>>(xg, xbf, bar);
  int tot = ESN_RT * ESN_KTP * 512;
  esn_swizzleW<<<(tot + 255) / 256, 256, 0, stream>>>(W, A);
  esn_persist<<<ESN_RT, 128, 0, stream>>>(A, Win, u, xg, xbf, bar, out);
}
